// FastAutoencoder_64553358459100
// MI455X (gfx1250) — matrix operand order repaired and checked by reading, not compiled, not run on hardware
//
#include <hip/hip_runtime.h>
#include <stdint.h>

#define B_ROWS 8192
#define D_MODEL 2048
#define N_DIRS 16384
#define K_TOP 32
#define AUXK 256

typedef unsigned short u16;
typedef __attribute__((ext_vector_type(16))) __bf16 bf16x16;
typedef __attribute__((ext_vector_type(8))) float floatx8;

union FragU { bf16x16 v; uint4 q[2]; };

__device__ __forceinline__ u16 f32_to_bf16(float f) {
  unsigned u = __float_as_uint(f);
  unsigned r = 0x7fffu + ((u >> 16) & 1u);
  return (u16)((u + r) >> 16);
}

// ---------------- prep: xc = x - pre_bias -> bf16 ----------------
__global__ __launch_bounds__(256) void conv_x_kernel(const float* __restrict__ x,
                                                     const float* __restrict__ pre_bias,
                                                     u16* __restrict__ xcb) {
  size_t i = (size_t)blockIdx.x * blockDim.x + threadIdx.x;
  int d = (int)(i & (D_MODEL - 1));
  xcb[i] = f32_to_bf16(x[i] - pre_bias[d]);
}

__global__ __launch_bounds__(256) void conv_w_kernel(const float* __restrict__ w,
                                                     u16* __restrict__ wb) {
  size_t i = (size_t)blockIdx.x * blockDim.x + threadIdx.x;
  wb[i] = f32_to_bf16(w[i]);
}

// ---------------- prep: W_dec [D, N] -> W_dec^T [N, D] ----------------
__global__ __launch_bounds__(1024) void transpose_kernel(const float* __restrict__ wdec,
                                                         float* __restrict__ wdecT) {
  __shared__ float tile[32][33];
  int n0 = blockIdx.x * 32;
  int d0 = blockIdx.y * 32;
  int tx = threadIdx.x, ty = threadIdx.y;
  tile[ty][tx] = wdec[(size_t)(d0 + ty) * N_DIRS + (n0 + tx)];
  __syncthreads();
  wdecT[(size_t)(n0 + ty) * D_MODEL + (d0 + tx)] = tile[tx][ty];
}

// ---------------- encoder GEMM: latents = xc @ W_enc^T + latent_bias ----------------
// Block = 256 thr = 8 waves; block tile 128(M)x128(N); wave tile 32(M)x64(N)
// = 2x4 grid of 16x16x32 bf16 WMMAs, K=2048 in steps of 32.
__global__ __launch_bounds__(256) void enc_gemm_kernel(const u16* __restrict__ xcb,
                                                       const u16* __restrict__ wencb,
                                                       const float* __restrict__ lbias,
                                                       float* __restrict__ latents,
                                                       int rowBase) {
  const int lane = threadIdx.x & 31;
  const int wave = threadIdx.x >> 5;
  const int r = lane & 15;
  const int hf = lane >> 4;
  const int blockM = blockIdx.x * 128 + (wave >> 1) * 32;
  const int blockN = blockIdx.y * 128 + (wave & 1) * 64;

  const floatx8 zero8 = {0.f, 0.f, 0.f, 0.f, 0.f, 0.f, 0.f, 0.f};
  floatx8 acc[2][4];
#pragma unroll
  for (int mt = 0; mt < 2; ++mt)
#pragma unroll
    for (int nt = 0; nt < 4; ++nt) acc[mt][nt] = zero8;

  const u16* aRow[2];
#pragma unroll
  for (int mt = 0; mt < 2; ++mt)
    aRow[mt] = xcb + (size_t)(rowBase + blockM + mt * 16 + r) * D_MODEL;
  const u16* bRow[4];
#pragma unroll
  for (int nt = 0; nt < 4; ++nt)
    bRow[nt] = wencb + (size_t)(blockN + nt * 16 + r) * D_MODEL;

  for (int kk = 0; kk < D_MODEL; kk += 32) {
    FragU a[2], b[4];
    const int ao0 = kk + hf * 8;        // A: lanes 0-15 k..k+7, lanes 16-31 k+8..k+15
    const int ao1 = kk + 16 + hf * 8;   //    then +16
    const int bo = kk + hf * 8;         // B: same k order as A
#pragma unroll
    for (int mt = 0; mt < 2; ++mt) {
      a[mt].q[0] = *(const uint4*)(aRow[mt] + ao0);
      a[mt].q[1] = *(const uint4*)(aRow[mt] + ao1);
    }
#pragma unroll
    for (int nt = 0; nt < 4; ++nt) {
      b[nt].q[0] = *(const uint4*)(bRow[nt] + bo);
      b[nt].q[1] = *(const uint4*)(bRow[nt] + bo + 16);
    }
#pragma unroll
    for (int mt = 0; mt < 2; ++mt)
#pragma unroll
      for (int nt = 0; nt < 4; ++nt)
        acc[mt][nt] = __builtin_amdgcn_wmma_f32_16x16x32_bf16(
            false, a[mt].v, false, b[nt].v, (short)0, acc[mt][nt], false, false);
  }

  // C/D layout: vgpr i -> M = i + 8*hf ; lane&15 -> N
#pragma unroll
  for (int nt = 0; nt < 4; ++nt) {
    const int col = blockN + nt * 16 + r;
    const float lb = lbias[col];
#pragma unroll
    for (int mt = 0; mt < 2; ++mt) {
      const int mbase = blockM + mt * 16 + hf * 8;
#pragma unroll
      for (int i = 0; i < 8; ++i)
        latents[(size_t)(mbase + i) * N_DIRS + col] = acc[mt][nt][i] + lb;
    }
  }
}

// ---------------- top-k per row: radix rank-select + bitonic sort ----------------
__global__ __launch_bounds__(256) void topk_kernel(const float* __restrict__ latents,
                                                   int* __restrict__ tki, float* __restrict__ tkv,
                                                   int* __restrict__ aki, float* __restrict__ akv,
                                                   int rowBase) {
  extern __shared__ unsigned char smem[];
  unsigned* skey = (unsigned*)smem;          // N_DIRS keys (order-preserving f32 map)
  int* hist = (int*)(skey + N_DIRS);         // 256
  unsigned* ck = (unsigned*)(hist + 256);    // 256 candidate keys
  int* ci = (int*)(ck + 256);                // 256 candidate indices
  int* scal = ci + 256;                      // [0]=bin [1]=want [2]=cnt_gt [3]=cnt_eq

  const int tid = threadIdx.x;
  const float* rowp = latents + (size_t)blockIdx.x * N_DIRS;

  for (int e = tid; e < N_DIRS; e += 256) {
    unsigned s = __float_as_uint(rowp[e]);
    skey[e] = (s & 0x80000000u) ? ~s : (s | 0x80000000u);
  }
  if (tid == 0) { scal[2] = 0; scal[3] = 0; }
  __syncthreads();

  // rank-AUXK threshold via 4x 8-bit radix passes (high byte first)
  unsigned prefix = 0;
  int want = AUXK;
  for (int shift = 24; shift >= 0; shift -= 8) {
    hist[tid] = 0;
    __syncthreads();
    const unsigned himask = (shift == 24) ? 0u : (0xFFFFFFFFu << (shift + 8));
    for (int e = tid; e < N_DIRS; e += 256) {
      unsigned key = skey[e];
      if ((key & himask) == prefix) atomicAdd(&hist[(key >> shift) & 255], 1);
    }
    __syncthreads();
    if (tid == 0) {
      int acc = 0, bin = 255;
      for (; bin > 0; --bin) {
        int c = hist[bin];
        if (acc + c >= want) break;
        acc += c;
      }
      scal[0] = bin;
      scal[1] = want - acc;
    }
    __syncthreads();
    prefix |= ((unsigned)scal[0]) << shift;
    want = scal[1];
    __syncthreads();
  }

  // compact: all keys > thr, plus `want` keys == thr -> exactly AUXK candidates
  const int neq = want;
  const int eqBase = AUXK - neq;
  for (int e = tid; e < N_DIRS; e += 256) {
    unsigned key = skey[e];
    if (key > prefix) {
      int p = atomicAdd(&scal[2], 1);
      if (p < AUXK) { ck[p] = key; ci[p] = e; }
    } else if (key == prefix) {
      int p = atomicAdd(&scal[3], 1);
      if (p < neq) { ck[eqBase + p] = key; ci[eqBase + p] = e; }
    }
  }
  __syncthreads();

  // bitonic sort 256 (key,idx) pairs, descending
  for (int ks = 2; ks <= AUXK; ks <<= 1) {
    for (int j = ks >> 1; j > 0; j >>= 1) {
      int p = tid ^ j;
      if (p > tid) {
        bool desc = ((tid & ks) == 0);
        unsigned ka = ck[tid], kb = ck[p];
        bool sw = desc ? (ka < kb) : (ka > kb);
        if (sw) {
          ck[tid] = kb; ck[p] = ka;
          int t = ci[tid]; ci[tid] = ci[p]; ci[p] = t;
        }
      }
      __syncthreads();
    }
  }

  const int grow = rowBase + blockIdx.x;
  unsigned key = ck[tid];
  unsigned s = (key & 0x80000000u) ? (key & 0x7fffffffu) : ~key;
  float v = __uint_as_float(s);
  float rv = fmaxf(v, 0.f);
  int ind = ci[tid];
  akv[(size_t)grow * AUXK + tid] = rv;
  aki[(size_t)grow * AUXK + tid] = ind;
  if (tid < K_TOP) {
    tkv[(size_t)grow * K_TOP + tid] = rv;
    tki[(size_t)grow * K_TOP + tid] = ind;
  }
}

// ---------------- sparse decode: recons = pre_bias + sum_k v * W_dec^T[idx] ----------------
__global__ __launch_bounds__(256) void decode_kernel(const float* __restrict__ tkv,
                                                     const int* __restrict__ tki,
                                                     const float* __restrict__ wdecT,
                                                     const float* __restrict__ pre_bias,
                                                     float* __restrict__ recons) {
  __shared__ float sv[K_TOP];
  __shared__ int si[K_TOP];
  const int row = blockIdx.x;
  const int tid = threadIdx.x;
  if (tid < K_TOP) {
    sv[tid] = tkv[(size_t)row * K_TOP + tid];
    si[tid] = tki[(size_t)row * K_TOP + tid];
  }
  __syncthreads();
  float acc[8];
#pragma unroll
  for (int i = 0; i < 8; ++i) acc[i] = pre_bias[tid + i * 256];
  for (int j = 0; j < K_TOP; ++j) {
    const float v = sv[j];
    const float* wr = wdecT + (size_t)si[j] * D_MODEL;
#pragma unroll
    for (int i = 0; i < 8; ++i) acc[i] = fmaf(v, wr[tid + i * 256], acc[i]);
  }
#pragma unroll
  for (int i = 0; i < 8; ++i)
    recons[(size_t)row * D_MODEL + tid + i * 256] = acc[i];
}

extern "C" void kernel_launch(void* const* d_in, const int* in_sizes, int n_in,
                              void* d_out, int out_size, void* d_ws, size_t ws_size,
                              hipStream_t stream) {
  const float* x = (const float*)d_in[0];
  const float* W_enc = (const float*)d_in[1];
  const float* W_dec = (const float*)d_in[2];
  const float* pre_bias = (const float*)d_in[3];
  const float* latent_bias = (const float*)d_in[4];
  (void)in_sizes; (void)n_in; (void)out_size;

  // outputs, flat in return order
  float* out = (float*)d_out;
  float* recons = out;
  size_t off = (size_t)B_ROWS * D_MODEL;
  int* tki = (int*)(out + off);          off += (size_t)B_ROWS * K_TOP;
  float* tkv = out + off;                off += (size_t)B_ROWS * K_TOP;
  int* aki = (int*)(out + off);          off += (size_t)B_ROWS * AUXK;
  float* akv = out + off;

  // workspace layout: wenc_bf16 (64MB) | wdecT (128MB) | xc_bf16 (32MB) | latents chunk
  char* ws = (char*)d_ws;
  u16* wencb = (u16*)ws;
  float* wdecT = (float*)(ws + ((size_t)64 << 20));
  u16* xcb = (u16*)(ws + ((size_t)192 << 20));
  float* latents = (float*)(ws + ((size_t)224 << 20));
  size_t latBytes = ws_size > ((size_t)224 << 20) ? ws_size - ((size_t)224 << 20) : 0;
  long cap = (long)(latBytes / ((size_t)N_DIRS * sizeof(float)));
  cap = (cap / 128) * 128;
  if (cap < 128) cap = 128;
  if (cap > B_ROWS) cap = B_ROWS;

  conv_x_kernel<<<(B_ROWS * D_MODEL) / 256, 256, 0, stream>>>(x, pre_bias, xcb);
  conv_w_kernel<<<(N_DIRS * D_MODEL) / 256, 256, 0, stream>>>(W_enc, wencb);
  transpose_kernel<<<dim3(N_DIRS / 32, D_MODEL / 32), dim3(32, 32), 0, stream>>>(W_dec, wdecT);

  const size_t shBytes = (size_t)N_DIRS * 4 + 256 * 4 * 3 + 64;
  for (int rb = 0; rb < B_ROWS; rb += (int)cap) {
    int rows = B_ROWS - rb;
    if (rows > cap) rows = (int)cap;
    enc_gemm_kernel<<<dim3(rows / 128, N_DIRS / 128), 256, 0, stream>>>(
        xcb, wencb, latent_bias, latents, rb);
    topk_kernel<<<rows, 256, shBytes, stream>>>(latents, tki, tkv, aki, akv, rb);
  }
  decode_kernel<<<B_ROWS, 256, 0, stream>>>(tkv, tki, wdecT, pre_bias, recons);
}
